// Head_67216238182821
// MI455X (gfx1250) — matrix operand order repaired and checked by reading, not compiled, not run on hardware
//
#include <hip/hip_runtime.h>

// B=8, T=2048, C=1024, H=64 single-head causal attention, fp32 I/O.
// MI455X (gfx1250, wave32) plan:
//   Kernel 0: convert Wq/Wk/Wv fp32 -> bf16 (tiny). Scale folded into Wq? No:
//             scale folded into q at projection store (kernel 1, mat==0).
//   Kernel 1: q/k/v projection. One wave = 16 rows x all 64 cols of one matrix;
//             x converted (v_cvt_pk_bf16_f32) once per 32-chunk, reused 4x.
//   Kernel 2: flash attention, 16 queries + full H=64 per wave. Computes
//             S^T = K*Q^T: softmax row stats = one ds_swizzle xor-16, and
//             exp(S^T) C-layout == A-operand layout of P(16x32) (no transpose).
//             Unmasked fast path for full tiles; masked path only on diagonal.
#define B_DIM 8
#define T_SEQ 2048
#define C_DIM 1024
#define H_DIM 64

typedef unsigned int u32;
typedef unsigned short u16;
typedef __attribute__((ext_vector_type(16))) __bf16 v16bf;
typedef __attribute__((ext_vector_type(2)))  __bf16 v2bf;
typedef __attribute__((ext_vector_type(8)))  float  v8f;
typedef __attribute__((ext_vector_type(4)))  float  v4f;
typedef __attribute__((ext_vector_type(8)))  u32    v8u;

// ---------- conversion helpers (native v_cvt_pk_bf16_f32 path) ----------
__device__ __forceinline__ u32 pk_bf16(float lo, float hi) {
  v2bf t;
  t[0] = (__bf16)lo;
  t[1] = (__bf16)hi;
  return __builtin_bit_cast(u32, t);
}
__device__ __forceinline__ u16 f32_bf16(float f) {
  return __builtin_bit_cast(u16, (__bf16)f);
}

__device__ __forceinline__ v8f wmma_bf16(v16bf a, v16bf b, v8f c) {
  return __builtin_amdgcn_wmma_f32_16x16x32_bf16(false, a, false, b, (short)0, c, false, false);
}

// A operand (16x32, 16-bit): lane&15 = row M; lanes<16 hold k {0-7,16-23},
// lanes>=16 hold k {8-15,24-31} (ISA 7.12.2), packed as bf16 pairs.
__device__ __forceinline__ v16bf ldA_bf16(const u16* base, int stride, int lane) {
  const u16* p = base + (size_t)(lane & 15) * stride + ((lane >> 4) << 3);
  v8u r;
#pragma unroll
  for (int j = 0; j < 8; ++j) {
    int kk = (j < 4) ? (2 * j) : (2 * j + 8);
    r[j] = *(const u32*)(p + kk);
  }
  return __builtin_bit_cast(v16bf, r);
}
__device__ __forceinline__ v16bf ldA_f32(const float* base, int stride, int lane) {
  const float* p = base + (size_t)(lane & 15) * stride + ((lane >> 4) << 3);
  v8u r;
#pragma unroll
  for (int j = 0; j < 8; ++j) {
    int kk = (j < 4) ? (2 * j) : (2 * j + 8);
    r[j] = pk_bf16(p[kk], p[kk + 1]);
  }
  return __builtin_bit_cast(v16bf, r);
}

// B operand (32x16, 16-bit): lane&15 = column N, elements = 16 contiguous
// k values of that column; caller passes the per-lane pointer.
__device__ __forceinline__ v16bf ldB_row_bf16(const u16* p) {
  v8u r;
#pragma unroll
  for (int j = 0; j < 8; ++j) r[j] = *(const u32*)(p + 2 * j);
  return __builtin_bit_cast(v16bf, r);
}
// B operand from column-strided source (X^T from row-major X).
__device__ __forceinline__ v16bf ldB_col_bf16(const u16* p, int cstride) {
  v8u r;
#pragma unroll
  for (int j = 0; j < 8; ++j)
    r[j] = ((u32)p[(size_t)(2 * j) * cstride]) |
           ((u32)p[(size_t)(2 * j + 1) * cstride] << 16);
  return __builtin_bit_cast(v16bf, r);
}
__device__ __forceinline__ v16bf ldB_tbl_bf16(const u16* p, int cstride) {
  v8u r;
#pragma unroll
  for (int j = 0; j < 8; ++j) {
    int kk = (j < 4) ? (2 * j) : (2 * j + 8);
    r[j] = ((u32)p[(size_t)kk * cstride]) |
           ((u32)p[(size_t)(kk + 1) * cstride] << 16);
  }
  return __builtin_bit_cast(v16bf, r);
}

__device__ __forceinline__ float swz_xor16(float v) {  // group-of-32 xor 0x10
  return __builtin_bit_cast(float, __builtin_amdgcn_ds_swizzle(__builtin_bit_cast(int, v), 0x401F));
}
__device__ __forceinline__ float lane_bcast(float v, int src) {
  return __builtin_bit_cast(float, __builtin_amdgcn_ds_bpermute(src << 2, __builtin_bit_cast(int, v)));
}

// ---------------- kernel 0: W fp32 -> bf16 (layout [mat][C][H]) ----------------
__global__ void wconv_kernel(const float* __restrict__ Wk, const float* __restrict__ Wq,
                             const float* __restrict__ Wv, u16* __restrict__ wb) {
  int idx = blockIdx.x * blockDim.x + threadIdx.x;       // one thread = 4 elems
  const int n4 = C_DIM * H_DIM / 4;                      // 16384 per matrix
  if (idx >= 3 * n4) return;
  int mat = idx / n4, e4 = (idx - mat * n4) * 4;
  const float* W = (mat == 0) ? Wq : (mat == 1) ? Wk : Wv;
  v4f f = *(const v4f*)(W + e4);
  u32 lo = pk_bf16(f[0], f[1]), hi = pk_bf16(f[2], f[3]);
  u32* o = (u32*)(wb + (size_t)mat * C_DIM * H_DIM + e4);
  o[0] = lo; o[1] = hi;
}

// ---------------- kernel 1: q/k/v projection ----------------
// waves: mat(3) x rowtile(B*T/16 = 1024) = 3072; each wave -> 16 x 64 tile.
__global__ void qkv_proj_kernel(const float* __restrict__ x,
                                const u16* __restrict__ wb,
                                u16* __restrict__ qo, u16* __restrict__ ko,
                                u16* __restrict__ vo) {
  int gw   = (blockIdx.x * blockDim.x + threadIdx.x) >> 5;
  int lane = threadIdx.x & 31;
  int rt   = gw & 1023;
  int mat  = gw >> 10;                                   // 0:q 1:k 2:v

  const u16* W = wb + (size_t)mat * C_DIM * H_DIM;
  u16* outp    = (mat == 0) ? qo : (mat == 1) ? ko : vo;
  const float* xa = x + (size_t)rt * 16 * C_DIM;

  v8f acc0 = {}, acc1 = {}, acc2 = {}, acc3 = {};
  for (int cc = 0; cc < C_DIM; cc += 32) {
    v16bf a = ldA_f32(xa + cc, C_DIM, lane);             // convert once...
    const u16* wrow = W + (size_t)(cc + ((lane >> 4) << 3)) * H_DIM + (lane & 15);   // ...reuse 4x
    acc0 = wmma_bf16(a, ldB_tbl_bf16(wrow, H_DIM),      acc0);
    acc1 = wmma_bf16(a, ldB_tbl_bf16(wrow + 16, H_DIM), acc1);
    acc2 = wmma_bf16(a, ldB_tbl_bf16(wrow + 32, H_DIM), acc2);
    acc3 = wmma_bf16(a, ldB_tbl_bf16(wrow + 48, H_DIM), acc3);
  }

  // Fold 1/sqrt(C) into q so the attention loop never multiplies by scale.
  const float os = (mat == 0) ? 0.03125f : 1.0f;
  int hi = lane >> 4, col = lane & 15;
  u16* o = outp + (size_t)rt * 16 * H_DIM + col;
#pragma unroll
  for (int r = 0; r < 8; ++r) {
    size_t row = (size_t)(r + (hi << 3)) * H_DIM;
    o[row]      = f32_bf16(acc0[r] * os);
    o[row + 16] = f32_bf16(acc1[r] * os);
    o[row + 32] = f32_bf16(acc2[r] * os);
    o[row + 48] = f32_bf16(acc3[r] * os);
  }
}

// ---------------- kernel 2: causal flash attention ----------------
// One 32-key tile step; MASKED=true only for the diagonal tile.
template <bool MASKED>
__device__ __forceinline__ void attn_step(
    int kb, int qglob, int lane, int hi,
    const u16* __restrict__ kp, const u16* __restrict__ vp,
    v16bf qB0, v16bf qB1,
    float& m_i, float& l_i,
    v8f& acc0, v8f& acc1, v8f& acc2, v8f& acc3) {
  v8f st0 = {}, st1 = {};
  st0 = wmma_bf16(ldA_bf16(kp + (size_t)kb * H_DIM, H_DIM, lane),             qB0, st0);
  st0 = wmma_bf16(ldA_bf16(kp + (size_t)kb * H_DIM + 32, H_DIM, lane),        qB1, st0);
  st1 = wmma_bf16(ldA_bf16(kp + (size_t)(kb + 16) * H_DIM, H_DIM, lane),      qB0, st1);
  st1 = wmma_bf16(ldA_bf16(kp + (size_t)(kb + 16) * H_DIM + 32, H_DIM, lane), qB1, st1);

  float p0[8], p1[8];
  float mloc = -1e30f;
#pragma unroll
  for (int r = 0; r < 8; ++r) {
    float s0 = st0[r], s1 = st1[r];
    if (MASKED) {
      int k0 = kb + r + (hi << 3);
      s0 = (k0 <= qglob)      ? s0 : -1e30f;
      s1 = (k0 + 16 <= qglob) ? s1 : -1e30f;
    }
    p0[r] = s0; p1[r] = s1;
    mloc = fmaxf(mloc, fmaxf(s0, s1));
  }
  mloc = fmaxf(mloc, swz_xor16(mloc));            // combine the two key halves
  float m_new = fmaxf(m_i, mloc);
  float alpha = __expf(m_i - m_new);
  float lsum = 0.0f;
#pragma unroll
  for (int r = 0; r < 8; ++r) {
    p0[r] = __expf(p0[r] - m_new);
    p1[r] = __expf(p1[r] - m_new);
    lsum += p0[r] + p1[r];
  }
  lsum += swz_xor16(lsum);
  l_i = l_i * alpha + lsum;
  m_i = m_new;

  // Rescale O accumulators (acc row m = r + 8*hi; alpha lives at lane m).
#pragma unroll
  for (int r = 0; r < 8; ++r) {
    float ar = lane_bcast(alpha, r + (hi << 3));
    acc0[r] *= ar; acc1[r] *= ar; acc2[r] *= ar; acc3[r] *= ar;
  }

  // P(16x32) as A operand: pure in-lane repack of the two S^T exp tiles.
  v8u pr;
#pragma unroll
  for (int j = 0; j < 4; ++j) {
    pr[j]     = pk_bf16(p0[2 * j], p0[2 * j + 1]);
    pr[j + 4] = pk_bf16(p1[2 * j], p1[2 * j + 1]);
  }
  v16bf pa = __builtin_bit_cast(v16bf, pr);

  const u16* vrow = vp + (size_t)(kb + (hi << 3)) * H_DIM + (lane & 15);
  acc0 = wmma_bf16(pa, ldB_tbl_bf16(vrow, H_DIM),      acc0);
  acc1 = wmma_bf16(pa, ldB_tbl_bf16(vrow + 16, H_DIM), acc1);
  acc2 = wmma_bf16(pa, ldB_tbl_bf16(vrow + 32, H_DIM), acc2);
  acc3 = wmma_bf16(pa, ldB_tbl_bf16(vrow + 48, H_DIM), acc3);
}

__global__ void flash_attn_kernel(const u16* __restrict__ q,
                                  const u16* __restrict__ k,
                                  const u16* __restrict__ v,
                                  float* __restrict__ out) {
  int gw   = (blockIdx.x * blockDim.x + threadIdx.x) >> 5;
  int lane = threadIdx.x & 31;
  int b    = gw >> 7;            // 128 query tiles per batch
  int qt   = gw & 127;
  int qb   = qt << 4;

  const u16* qp = q + ((size_t)b * T_SEQ + qb) * H_DIM;
  const u16* kp = k + (size_t)b * T_SEQ * H_DIM;
  const u16* vp = v + (size_t)b * T_SEQ * H_DIM;

  v16bf qB0 = ldA_bf16(qp, H_DIM, lane);       // Q^T cols = lane&15, h 0..31
  v16bf qB1 = ldA_bf16(qp + 32, H_DIM, lane);  // Q^T cols = lane&15, h 32..63

  v8f acc0 = {}, acc1 = {}, acc2 = {}, acc3 = {};
  float m_i = -1e30f, l_i = 0.0f;
  int hi = lane >> 4, col = lane & 15;
  int qglob = qb + col;

  // Full (never-masked) tiles: keys kb..kb+31 all <= every query in the slab.
  int kdiag = qb & ~31;
  for (int kb = 0; kb < kdiag; kb += 32)
    attn_step<false>(kb, qglob, lane, hi, kp, vp, qB0, qB1,
                     m_i, l_i, acc0, acc1, acc2, acc3);
  // Diagonal tile: partial causal mask.
  attn_step<true>(kdiag, qglob, lane, hi, kp, vp, qB0, qB1,
                  m_i, l_i, acc0, acc1, acc2, acc3);

  float* op = out + ((size_t)b * T_SEQ + qb) * H_DIM;
#pragma unroll
  for (int r = 0; r < 8; ++r) {
    float lr  = lane_bcast(l_i, r + (hi << 3));
    float inv = (lr > 0.0f) ? 1.0f / lr : 0.0f;
    int row = r + (hi << 3);
    op[(size_t)row * H_DIM + col]      = acc0[r] * inv;
    op[(size_t)row * H_DIM + col + 16] = acc1[r] * inv;
    op[(size_t)row * H_DIM + col + 32] = acc2[r] * inv;
    op[(size_t)row * H_DIM + col + 48] = acc3[r] * inv;
  }
}

extern "C" void kernel_launch(void* const* d_in, const int* in_sizes, int n_in,
                              void* d_out, int out_size, void* d_ws, size_t ws_size,
                              hipStream_t stream) {
  (void)in_sizes; (void)n_in; (void)out_size; (void)ws_size;
  const float* x  = (const float*)d_in[0];
  const float* Wk = (const float*)d_in[1];
  const float* Wq = (const float*)d_in[2];
  const float* Wv = (const float*)d_in[3];

  const size_t nTH = (size_t)B_DIM * T_SEQ * H_DIM;      // 1M elems
  u16* qbuf = (u16*)d_ws;
  u16* kbuf = qbuf + nTH;
  u16* vbuf = kbuf + nTH;
  u16* wbuf = vbuf + nTH;                                // ~6.8 MB total

  wconv_kernel<<<192, 256, 0, stream>>>(Wk, Wq, Wv, wbuf);
  qkv_proj_kernel<<<384, 256, 0, stream>>>(x, wbuf, qbuf, kbuf, vbuf);
  flash_attn_kernel<<<128, 256, 0, stream>>>(qbuf, kbuf, vbuf, (float*)d_out);
}
